// QuantLinear_50379966382509
// MI455X (gfx1250) — matrix operand order repaired and checked by reading, not compiled, not run on hardware
//
#include <hip/hip_runtime.h>
#include <stdint.h>

// ---------------- problem constants (fixed by the reference) ----------------
#define MDIM 2048
#define KDIM 4096
#define NDIM 8192
#define GROUPS 32          // K / 128
#define GROUP_SIZE 128
#define BM 128
#define BN 256
#define BK 32              // one v_wmma_f32_16x16x32_f16 K-slab per step
#define NT (KDIM / BK)     // 128 K-steps
#define LDS_STRIDE 40      // halves per row (80B): 16B-aligned, conflict-free banks

typedef _Float16 v16h __attribute__((ext_vector_type(16)));
typedef _Float16 v8h  __attribute__((ext_vector_type(8)));
typedef _Float16 v2h  __attribute__((ext_vector_type(2)));
typedef float    v8f  __attribute__((ext_vector_type(8)));
typedef int      v4i  __attribute__((ext_vector_type(4)));

union H16 { v16h v; v8h h[2]; };
union B8  { v8h  v; v2h p[4]; };
union U32H2 { unsigned u; v2h h; };

__global__ __launch_bounds__(256)
void w4a16_wmma_gemm(const _Float16* __restrict__ A,      // [M,K] fp16
                     const int*      __restrict__ qw,     // [N,K/2] int32, 2 nibbles/low byte
                     const _Float16* __restrict__ scales, // [N,G]
                     const _Float16* __restrict__ zeros,  // [N,G]
                     const _Float16* __restrict__ bias,   // [N]
                     _Float16*       __restrict__ out)    // [M,N]
{
    __shared__ _Float16 lsA[2][BM * LDS_STRIDE];   // activation tile (async DMA target)
    __shared__ _Float16 lsB[2][BN * LDS_STRIDE];   // dequantized weight tile

    const int tid   = threadIdx.x;
    const int lane  = tid & 31;          // wave32
    const int wid   = tid >> 5;          // 8 waves
    const int waveM = wid >> 2;          // 0..1  -> 64 rows each
    const int waveN = wid & 3;           // 0..3  -> 64 cols each

    const int blockN = blockIdx.x;       // 32 blocks
    const int blockM = blockIdx.y;       // 16 blocks

    // ---------------- loader addressing ----------------
    // A: 2 threads per row (128 rows), async DMA global->LDS, 32B each
    const int  lrow  = tid >> 1;                          // 0..127
    const int  lhalf = tid & 1;                           // which 16-K half
    const long aRowBase = (long)(blockM * BM + lrow) * KDIM;      // elements
    // B: 1 thread per row (256 rows), 16 int32 (= 32 nibbles = one BK slab) each
    const long qRowBase = (long)(blockN * BN + tid) * (KDIM / 2);
    const int  sRowBase = (blockN * BN + tid) * GROUPS;

    v4i  rq0, rq1, rq2, rq3;  // staged qweight (16 int32 = 32 nibbles)
    _Float16 rs, rz;          // staged scale / zero for this group

    // A tile: pure copy -> CDNA5 async global->LDS DMA (ASYNCcnt), 2x b128 per thread
    auto issue_async_A = [&](int kt, int buf) {
        const unsigned lds0 =
            (unsigned)(uintptr_t)&lsA[buf][lrow * LDS_STRIDE + lhalf * 16];
        const unsigned goff =
            (unsigned)(2L * (aRowBase + (long)kt * BK + lhalf * 16)); // byte offset
        asm volatile(
            "global_load_async_to_lds_b128 %0, %1, %2\n\t"
            "global_load_async_to_lds_b128 %0, %1, %2 offset:16"
            :
            : "v"(lds0), "v"(goff), "s"(A)
            : "memory");
    };

    auto load_B = [&](int kt) {
        const int* qp = qw + qRowBase + kt * (BK / 2);
        rq0 = *reinterpret_cast<const v4i*>(qp);
        rq1 = *reinterpret_cast<const v4i*>(qp + 4);
        rq2 = *reinterpret_cast<const v4i*>(qp + 8);
        rq3 = *reinterpret_cast<const v4i*>(qp + 12);
        const int g = (kt * BK) / GROUP_SIZE;           // whole slab in one group
        rs = scales[sRowBase + g];
        rz = zeros [sRowBase + g];
    };

    // dequant via fp16 bit-trick: 0x6400|v == fp16(1024+v); both nibbles packed,
    // then packed-fp16 (w - (1024+z)) * s  ->  v_pk_add_f16(neg) / v_pk_mul_f16
    auto dequant8 = [](v4i q, v2h k2, v2h s2) -> v8h {
        const int qq[4] = { q.x, q.y, q.z, q.w };
        B8 w;
        #pragma unroll
        for (int j = 0; j < 4; ++j) {
            const unsigned u = (unsigned)qq[j];
            U32H2 t;
            t.u = 0x64006400u | (u & 0xFu) | ((u & 0xF0u) << 12);
            w.p[j] = (t.h - k2) * s2;
        }
        return w.v;
    };

    auto store_B = [&](int buf) {
        const _Float16 kz = (_Float16)1024.0f + rz;
        const v2h k2 = {kz, kz};
        const v2h s2 = {rs, rs};
        _Float16* lb = &lsB[buf][tid * LDS_STRIDE];
        *reinterpret_cast<v8h*>(lb)      = dequant8(rq0, k2, s2);   // K 0..7
        *reinterpret_cast<v8h*>(lb + 8)  = dequant8(rq1, k2, s2);   // K 8..15
        *reinterpret_cast<v8h*>(lb + 16) = dequant8(rq2, k2, s2);   // K 16..23
        *reinterpret_cast<v8h*>(lb + 24) = dequant8(rq3, k2, s2);   // K 24..31
    };

    // ---------------- WMMA fragment addressing ----------------
    const int halfSel = lane >> 4;       // 0: lanes 0-15, 1: lanes 16-31
    const int l15     = lane & 15;
    const int aK0     = halfSel * 8;     // A: lanes16-31 hold K {8-15,24-31}
    const int bK0     = halfSel * 8;     // B: lanes16-31 hold K {8-15,24-31}

    v8f acc[4][4] = {};                  // 64x64 of C per wave

    auto compute = [&](int buf) {
        H16 afrag[4], bfrag[4];
        #pragma unroll
        for (int mi = 0; mi < 4; ++mi) {
            const _Float16* p =
                &lsA[buf][(waveM * 64 + mi * 16 + l15) * LDS_STRIDE + aK0];
            afrag[mi].h[0] = *reinterpret_cast<const v8h*>(p);        // K base..+7
            afrag[mi].h[1] = *reinterpret_cast<const v8h*>(p + 16);   // K base+16..+23
        }
        #pragma unroll
        for (int ni = 0; ni < 4; ++ni) {
            const _Float16* p =
                &lsB[buf][(waveN * 64 + ni * 16 + l15) * LDS_STRIDE + bK0];
            bfrag[ni].h[0] = *reinterpret_cast<const v8h*>(p);        // K base..+7
            bfrag[ni].h[1] = *reinterpret_cast<const v8h*>(p + 16);   // K base+16..+23
        }
        #pragma unroll
        for (int mi = 0; mi < 4; ++mi)
            #pragma unroll
            for (int ni = 0; ni < 4; ++ni)
                acc[mi][ni] = __builtin_amdgcn_wmma_f32_16x16x32_f16(
                    /*neg_a=*/false, afrag[mi].v,
                    /*neg_b=*/false, bfrag[ni].v,
                    /*c_mod=*/(short)0, acc[mi][ni],
                    /*reuse_a=*/false, /*reuse_b=*/false);
    };

    // ---------------- pipelined main loop ----------------
    issue_async_A(0, 0);
    load_B(0);
    store_B(0);
    asm volatile("s_wait_asynccnt 0x0" ::: "memory");   // A tile 0 landed in LDS
    __syncthreads();

    for (int kt = 0; kt < NT; ++kt) {
        const int cur = kt & 1;
        if (kt + 1 < NT) {
            issue_async_A(kt + 1, cur ^ 1);  // DMA next A tile while we compute
            load_B(kt + 1);                  // global -> regs overlaps WMMA
        }
        if (kt + 2 < NT)                     // warm L2 for k+2 weights
            __builtin_prefetch(qw + qRowBase + (kt + 2) * (BK / 2), 0, 3);

        compute(cur);

        if (kt + 1 < NT) {
            asm volatile("s_wait_asynccnt 0x0" ::: "memory"); // our async writes done
            __syncthreads();                 // all waves done reading `cur` + async done
            store_B(cur ^ 1);
            __syncthreads();
        }
    }

    // ---------------- epilogue: C(f32) + bias -> fp16 ----------------
    // C layout: VGPR r -> row (r + 8*halfSel), lane&15 -> col
    #pragma unroll
    for (int ni = 0; ni < 4; ++ni) {
        const int col = blockN * BN + waveN * 64 + ni * 16 + l15;
        const float bv = (float)bias[col];
        #pragma unroll
        for (int mi = 0; mi < 4; ++mi) {
            const int row0 = blockM * BM + waveM * 64 + mi * 16 + halfSel * 8;
            #pragma unroll
            for (int r = 0; r < 8; ++r)
                out[(long)(row0 + r) * NDIM + col] = (_Float16)(acc[mi][ni][r] + bv);
        }
    }
}

extern "C" void kernel_launch(void* const* d_in, const int* in_sizes, int n_in,
                              void* d_out, int out_size, void* d_ws, size_t ws_size,
                              hipStream_t stream) {
    (void)in_sizes; (void)n_in; (void)d_ws; (void)ws_size; (void)out_size;
    const _Float16* A      = (const _Float16*)d_in[0];
    const int*      qw     = (const int*)d_in[1];
    const _Float16* scales = (const _Float16*)d_in[2];
    const _Float16* zeros  = (const _Float16*)d_in[3];
    const _Float16* bias   = (const _Float16*)d_in[4];
    _Float16*       out    = (_Float16*)d_out;

    dim3 grid(NDIM / BN, MDIM / BM);   // 32 x 16 blocks
    dim3 block(256);                   // 8 wave32 waves
    w4a16_wmma_gemm<<<grid, block, 0, stream>>>(A, qw, scales, zeros, bias, out);
}
